// QattenMixer_28149215658049
// MI455X (gfx1250) — matrix operand order repaired and checked by reading, not compiled, not run on hardware
//
#include <hip/hip_runtime.h>

typedef _Float16 v16h __attribute__((ext_vector_type(16)));
typedef _Float16 v8h  __attribute__((ext_vector_type(8)));
typedef float    v8f  __attribute__((ext_vector_type(8)));

#define BROWS 32768
#define NTILES 22      // 352/16
#define KSTEPS 20      // 640/32

// ---- ws layout (in halfs) ----
#define OFF_W1      0
#define SZ_W1       (NTILES*KSTEPS*512)        // 225280
#define OFF_SELW2   (OFF_W1 + SZ_W1)
#define SZ_SELW2    (4*2*2*512)                // 8192
#define OFF_WHW2    (OFF_SELW2 + SZ_SELW2)
#define SZ_WHW2     (2*512)
#define OFF_VW2     (OFF_WHW2 + SZ_WHW2)
#define SZ_VW2      512
#define OFF_KEYW    (OFF_VW2 + SZ_VW2)
#define SZ_KEYW     (4*512)
#define HALFS_TOTAL (OFF_KEYW + SZ_KEYW)       // 237056 halfs
#define BIAS_N      352
#define PACK_TOTAL  (HALFS_TOTAL + BIAS_N)     // ws bytes needed: ~475.5 KB

// ---- LDS layout (bytes) ----
#define SP   648    // S pitch in halfs (row = 1296B = 81*16, spreads banks)
#define HP   360    // hidden pitch halfs (720B, 16B aligned)
#define SELP 136    // sel pitch halfs (272B)
#define SUP  68     // selU pitch floats
#define LDS_S    0
#define LDS_HID  (64*SP*2)                 // 82944
#define LDS_SEL  (LDS_HID + 64*HP*2)       // 129024
#define LDS_WHV  (LDS_SEL + 64*SELP*2)     // 146432
#define LDS_Q    (LDS_WHV + 64*8*4)        // 148480
#define LDS_RED  (LDS_Q + 64*32*4)         // 156672
#define LDS_ACC  (LDS_RED + 64*4*4)        // 157696
#define LDS_TOTAL (LDS_ACC + 8*4)          // 157728 bytes (<320KB WGP LDS)

// Pack all weights into WMMA B-fragment layout (f16) + concat bias (f32).
// B operand, contiguous k order used here (K=32 step): lane l = n + 16*g, half h -> K = 16*g + h.
__global__ void qatten_pack(const float* selW1, const float* selb1, const float* selW2,
                            const float* keyW, const float* whW1, const float* whb1,
                            const float* whW2, const float* vW1, const float* vb1,
                            const float* vW2, _Float16* ws) {
  int idx = blockIdx.x * 256 + threadIdx.x;
  if (idx >= PACK_TOTAL) return;
  if (idx >= HALFS_TOTAL) {                       // concat bias as f32
    int n = idx - HALFS_TOTAL;
    float v = (n < 256) ? selb1[n] : (n < 320) ? whb1[n - 256] : vb1[n - 320];
    ((float*)(ws + HALFS_TOTAL))[n] = v;
    return;
  }
  int off = idx & 511, lane = off >> 4, h = off & 15;
  int nl = lane & 15, g = lane >> 4;
  float v = 0.0f;
  if (idx < OFF_SELW2) {                          // W1cat: [352 out][640 in]
    int tile = idx >> 9;
    int nt = tile / KSTEPS, ks = tile % KSTEPS;
    int n = nt * 16 + nl;
    int k = ks * 32 + 16 * g + h;
    if (n < 256)      v = selW1[n * 640 + k];     // sel_W1 flat [4][64][640]
    else if (n < 320) v = whW1[(n - 256) * 640 + k];
    else              v = vW1[(n - 320) * 640 + k];
    ws[OFF_W1 + idx] = (_Float16)v; return;
  }
  if (idx < OFF_WHW2) {                           // sel_W2: per head [32 out][64 in]
    int j = idx - OFF_SELW2; int blk = j >> 9;
    int hh = blk >> 2, nt = (blk >> 1) & 1, ks = blk & 1;
    int d = nt * 16 + nl, k = ks * 32 + 16 * g + h;
    v = selW2[(hh * 32 + d) * 64 + k];
    ws[idx] = (_Float16)v; return;
  }
  if (idx < OFF_VW2) {                            // wh_W2: [4 out][64 in], pad n>=4 with 0
    int j = idx - OFF_WHW2; int ks = j >> 9;
    int k = ks * 32 + 16 * g + h;
    v = (nl < 4) ? whW2[nl * 64 + k] : 0.0f;
    ws[idx] = (_Float16)v; return;
  }
  if (idx < OFF_KEYW) {                           // v_W2: [1 out][32 in]
    int k = 16 * g + h;
    v = (nl == 0) ? vW2[k] : 0.0f;
    ws[idx] = (_Float16)v; return;
  }
  { // key_W^T for selU: B[k=d][n=u] = key_W[h][d][u]; K=32, out=16
    int j = idx - OFF_KEYW; int hh = j >> 9;
    int k = 16 * g + h;                           // d index, 0..31
    v = keyW[hh * 512 + k * 16 + nl];
    ws[idx] = (_Float16)v;
  }
}

__global__ __launch_bounds__(256)
void qatten_main(const float* __restrict__ states, const float* __restrict__ agent_qs,
                 const float* __restrict__ whb2, const float* __restrict__ vb2,
                 const _Float16* __restrict__ ws, float* __restrict__ out) {
  extern __shared__ char smem[];
  _Float16* S   = (_Float16*)(smem + LDS_S);
  _Float16* H   = (_Float16*)(smem + LDS_HID);
  float*    SU  = (float*)(smem + LDS_HID);       // alias: hidden dead before selU written
  _Float16* SEL = (_Float16*)(smem + LDS_SEL);
  float*    WHV = (float*)(smem + LDS_WHV);
  float*    Q   = (float*)(smem + LDS_Q);
  float*    RED = (float*)(smem + LDS_RED);
  float*    ACC = (float*)(smem + LDS_ACC);

  const int tid = threadIdx.x;
  const int wave = tid >> 5, lane = tid & 31;
  const int lm = lane & 15, lg = lane >> 4;
  const int rbase = blockIdx.x * 64;
  const float* bias = (const float*)(ws + HALFS_TOTAL);

  // Phase 1: stage S block [64][640] f32->f16 and q block into LDS
  for (int i = tid; i < 64 * 640; i += 256) {
    int r = i / 640, c = i - r * 640;
    S[r * SP + c] = (_Float16)states[(size_t)(rbase + r) * 640 + c];
  }
  for (int i = tid; i < 64 * 32; i += 256)
    Q[i] = agent_qs[(size_t)rbase * 32 + i];
  if (tid < 5) ACC[tid] = 0.0f;
  __syncthreads();

  // Phase 2: GEMM1  hidden[64][352] = relu(S * W1cat^T + bias)
  // A operand 16x32, same contiguous k order as packed B: lane = m + 16g, half h -> K = 16*g + h
  for (int t = wave; t < 88; t += 8) {
    int mt = t & 3, nt = t >> 2;
    v8f acc = {};
    const _Float16* arow = S + (mt * 16 + lm) * SP;
    for (int ks = 0; ks < KSTEPS; ++ks) {
      union { v16h v; v8h h[2]; } a;
      int ca = ks * 32 + 16 * lg;
      a.h[0] = *(const v8h*)(arow + ca);
      a.h[1] = *(const v8h*)(arow + ca + 8);
      v16h b = *(const v16h*)(ws + OFF_W1 + (size_t)(nt * KSTEPS + ks) * 512 + lane * 16);
      acc = __builtin_amdgcn_wmma_f32_16x16x32_f16(false, a.v, false, b, (short)0, acc, false, false);
    }
    int col = nt * 16 + lm;
    float bz = bias[col];
#pragma unroll
    for (int r = 0; r < 8; ++r) {
      float v = acc[r] + bz;
      v = v > 0.0f ? v : 0.0f;
      H[(mt * 16 + r + 8 * lg) * HP + col] = (_Float16)v;
    }
  }
  __syncthreads();

  // Phase 3a: sel (4 heads, K=64->32), w_head raw (K=64->4), v raw (K=32->1)
  for (int t = wave; t < 40; t += 8) {
    int mt, acol, nks, mode, hh = 0, nt2 = 0;
    const _Float16* bp;
    if (t < 32)      { hh = t >> 3; int rem = t & 7; mt = rem & 3; nt2 = rem >> 2;
                       acol = hh * 64; nks = 2; bp = ws + OFF_SELW2 + (hh * 2 + nt2) * 2 * 512; mode = 0; }
    else if (t < 36) { mt = t - 32; acol = 256; nks = 2; bp = ws + OFF_WHW2; mode = 1; }
    else             { mt = t - 36; acol = 320; nks = 1; bp = ws + OFF_VW2;  mode = 2; }
    v8f acc = {};
    const _Float16* arow = H + (mt * 16 + lm) * HP + acol;
    for (int ks = 0; ks < nks; ++ks) {
      union { v16h v; v8h h[2]; } a;
      int ca = ks * 32 + 16 * lg;
      a.h[0] = *(const v8h*)(arow + ca);
      a.h[1] = *(const v8h*)(arow + ca + 8);
      v16h b = *(const v16h*)(bp + ks * 512 + lane * 16);
      acc = __builtin_amdgcn_wmma_f32_16x16x32_f16(false, a.v, false, b, (short)0, acc, false, false);
    }
    if (mode == 0) {
      int d = nt2 * 16 + lm;
#pragma unroll
      for (int r = 0; r < 8; ++r)
        SEL[(mt * 16 + r + 8 * lg) * SELP + hh * 32 + d] = (_Float16)acc[r];
    } else if (mode == 1) {
      if (lm < 4)
#pragma unroll
        for (int r = 0; r < 8; ++r) WHV[(mt * 16 + r + 8 * lg) * 8 + lm] = acc[r];
    } else {
      if (lm == 0)
#pragma unroll
        for (int r = 0; r < 8; ++r) WHV[(mt * 16 + r + 8 * lg) * 8 + 4] = acc[r];
    }
  }
  __syncthreads();

  // Phase 3b: selU[h,b,:16] = sel[h,b,:32] * key_W[h]  (K=32)
  for (int t = wave; t < 16; t += 8) {
    int hh = t >> 2, mt = t & 3;
    v8f acc = {};
    const _Float16* arow = SEL + (mt * 16 + lm) * SELP + hh * 32;
    union { v16h v; v8h h[2]; } a;
    a.h[0] = *(const v8h*)(arow + 16 * lg);
    a.h[1] = *(const v8h*)(arow + 16 * lg + 8);
    v16h b = *(const v16h*)(ws + OFF_KEYW + hh * 512 + lane * 16);
    acc = __builtin_amdgcn_wmma_f32_16x16x32_f16(false, a.v, false, b, (short)0, acc, false, false);
#pragma unroll
    for (int r = 0; r < 8; ++r)
      SU[(mt * 16 + r + 8 * lg) * SUP + hh * 16 + lm] = acc[r];
  }
  __syncthreads();

  // Phase 4: per (row, head) — logits, softmax, head_q, outputs + reductions
  {
    int row = tid >> 2, head = tid & 3;
    const float* su = SU + row * SUP + head * 16;
    float sel_u[16];
#pragma unroll
    for (int u = 0; u < 16; ++u) sel_u[u] = su[u];
    const _Float16* unit = S + row * SP;    // unit = S[:, :512]
    float l[32], sq = 0.0f;
#pragma unroll
    for (int a = 0; a < 32; ++a) {
      float s = 0.0f;
#pragma unroll
      for (int u = 0; u < 16; ++u) s += (float)unit[a * 16 + u] * sel_u[u];
      l[a] = s; sq += s * s;
    }
    const float scale = 0.17677669529663687f;   // 1/sqrt(32)
    float mx = -1e30f;
#pragma unroll
    for (int a = 0; a < 32; ++a) { l[a] *= scale; mx = fmaxf(mx, l[a]); }
    float Z = 0.0f;
#pragma unroll
    for (int a = 0; a < 32; ++a) { l[a] = expf(l[a] - mx); Z += l[a]; }
    float inv = 1.0f / Z, ent = 0.0f, hq = 0.0f;
#pragma unroll
    for (int a = 0; a < 32; ++a) {
      float w = l[a] * inv;
      ent += w * logf(w + 1e-8f);
      hq += Q[row * 32 + a] * w;
    }
    float whr = WHV[row * 8 + head] + whb2[head];
    RED[row * 4 + head] = fabsf(whr) * hq;
    atomicAdd(&ACC[head], ent);
    atomicAdd(&ACC[4], sq);
    __syncthreads();
    if (head == 0) {
      float y = RED[row * 4] + RED[row * 4 + 1] + RED[row * 4 + 2] + RED[row * 4 + 3]
              + WHV[row * 8 + 4] + vb2[0];
      out[rbase + row] = y;
    }
    if (tid < 4) atomicAdd(&out[BROWS + 1 + tid], -ACC[tid] * (1.0f / 32768.0f));
    if (tid == 4) atomicAdd(&out[BROWS], ACC[4] * (1e-3f / (32768.0f * 32.0f)));
  }
}

extern "C" void kernel_launch(void* const* d_in, const int* in_sizes, int n_in,
                              void* d_out, int out_size, void* d_ws, size_t ws_size,
                              hipStream_t stream) {
  const float* agent_qs = (const float*)d_in[0];
  const float* states   = (const float*)d_in[1];
  // d_in[2] = actions (int) unused (mask_dead=False)
  const float* selW1 = (const float*)d_in[3];
  const float* selb1 = (const float*)d_in[4];
  const float* selW2 = (const float*)d_in[5];
  const float* keyW  = (const float*)d_in[6];
  const float* whW1  = (const float*)d_in[7];
  const float* whb1  = (const float*)d_in[8];
  const float* whW2  = (const float*)d_in[9];
  const float* whb2  = (const float*)d_in[10];
  const float* vW1   = (const float*)d_in[11];
  const float* vb1   = (const float*)d_in[12];
  const float* vW2   = (const float*)d_in[13];
  const float* vb2   = (const float*)d_in[14];
  _Float16* ws = (_Float16*)d_ws;
  float* out = (float*)d_out;

  // zero the scalar reduction outputs (attend_mag + 4 entropies)
  hipMemsetAsync(out + BROWS, 0, 5 * sizeof(float), stream);
  qatten_pack<<<(PACK_TOTAL + 255) / 256, 256, 0, stream>>>(
      selW1, selb1, selW2, keyW, whW1, whb1, whW2, vW1, vb1, vW2, ws);
  qatten_main<<<BROWS / 64, 256, LDS_TOTAL, stream>>>(states, agent_qs, whb2, vb2, ws, out);
}
